// LinearOperator2d_16123307229867
// MI455X (gfx1250) — matrix operand order repaired and checked by reading, not compiled, not run on hardware
//
#include <hip/hip_runtime.h>

typedef __attribute__((ext_vector_type(16))) _Float16 v16h;
typedef __attribute__((ext_vector_type(8)))  _Float16 v8h;
typedef __attribute__((ext_vector_type(4)))  _Float16 v4h;
typedef __attribute__((ext_vector_type(8)))  float    v8f;

#define HID_PAD   296                    // 288 K-half + 8 pad (f16 per c-row) -> 592B row, bank-spread
#define HID_BYTES (64 * HID_PAD * 2)     // 37888 B
#define SMEM_BYTES 65536                 // union: phase1 hid tile (+ty/tx), phase2 M (256x64 f32)

// ---------------------------------------------------------------------------
// Prep: f32 -> f16 copy of v as (256 = b*Ci rows, 576 = H*W cols), plus
// S[bi] = sum_HW v[b,i,H,W] for the b2 bias-through-einsum term.
// ---------------------------------------------------------------------------
__global__ __launch_bounds__(256) void prep_kernel(const float* __restrict__ v,
                                                   _Float16* __restrict__ vh,
                                                   float* __restrict__ S) {
  const int t = threadIdx.x;                       // row bi in [0,256)
  const float4* vr = (const float4*)(v + t * 576);
  v4h* dst = (v4h*)(vh + t * 576);
  float s = 0.f;
#pragma unroll 4
  for (int q = 0; q < 144; ++q) {
    float4 x = vr[q];
    s += x.x + x.y + x.z + x.w;
    v4h h;
    h[0] = (_Float16)x.x; h[1] = (_Float16)x.y;
    h[2] = (_Float16)x.z; h[3] = (_Float16)x.w;
    dst[q] = h;
  }
  S[t] = s;
}

// ---------------------------------------------------------------------------
// Main fused kernel: one block per output spatial position hw = (h,w).
// Phase 1: M(256x64) = V2(256x576) @ hid_hw(576x64) via v_wmma_f32_16x16x32_f16,
//          hid generated on the fly into LDS (two K-halves of 288).
// Phase 2: out[b,o,hw] = (1/576)*( sum_{i,c} M[b*16+i,c]*w2[o*16+i,c]
//                                  + sum_i b2[o*16+i]*S[b*16+i] ) + bias[o]
// ---------------------------------------------------------------------------
__global__ __launch_bounds__(256) void lin2d_kernel(
    const _Float16* __restrict__ vh, const float* __restrict__ S,
    const float* __restrict__ w1, const float* __restrict__ b1,
    const float* __restrict__ w2, const float* __restrict__ b2,
    const float* __restrict__ bias, float* __restrict__ out) {
  __shared__ __attribute__((aligned(16))) char smem[SMEM_BYTES];
  _Float16* hidT = (_Float16*)smem;                // phase 1: [c][k] f16, row stride HID_PAD
  float* ty = (float*)(smem + HID_BYTES);          // 24 floats
  float* tx = ty + 24;                             // 24 floats
  float* Mlds = (float*)smem;                      // phase 2: [bi][c] f32 (256x64)

  const int t    = threadIdx.x;
  const int lane = t & 31;
  const int wv   = t >> 5;                         // wave id 0..7
  const int hw   = blockIdx.x;                     // 0..575
  const int hh   = hw / 24;
  const int ww   = hw - hh * 24;

  // ty[Y] = tanh(2*(cy[Y]-cy[h])), tx[X] = tanh(2*(cx[X]-cx[w]))
  if (t < 48) {
    const int q    = (t < 24) ? t : (t - 24);
    const int base = (t < 24) ? hh : ww;
    const float cq = (2.0f / 24.0f) * ((float)q + 0.5f) - 1.0f;
    const float cb = (2.0f / 24.0f) * ((float)base + 0.5f) - 1.0f;
    const float val = tanhf(2.0f * (cq - cb));
    if (t < 24) ty[q] = val; else tx[q] = val;
  }

  v8f acc[2][4];
#pragma unroll
  for (int mt = 0; mt < 2; ++mt)
#pragma unroll
    for (int nt = 0; nt < 4; ++nt)
#pragma unroll
      for (int e = 0; e < 8; ++e) acc[mt][nt][e] = 0.0f;

  // hid generation work split: thread owns one c-row, 72 k's per half
  const int   hc   = t >> 2;                       // c in [0,64)
  const float w1a  = w1[2 * hc];
  const float w1b  = w1[2 * hc + 1];
  const float b1c  = b1[hc];
  const int   ks0  = (t & 3) * 72;

  // WMMA fragment addressing (wave32 16-bit layouts)
  const int row0   = wv * 32 + (lane & 15);        // A: M-row of v2
  const int colSel = (lane < 16) ? 0 : 8;          // A: lanes16-31 take K {8..15,24..31}
  const int kgsel  = (lane >> 4) << 3;             // B: lanes16-31 take K {8..15,24..31}

  for (int half = 0; half < 2; ++half) {
    const int kbase = half * 288;
    __syncthreads();                               // hid tile free to rewrite
    {
      _Float16* hrow = hidT + hc * HID_PAD;
#pragma unroll 4
      for (int j = 0; j < 72; ++j) {
        const int kl = ks0 + j;                    // local k in [0,288)
        const int k  = kbase + kl;                 // global HW index
        const int Y  = k / 24;
        const int X  = k - Y * 24;
        const float pre = ty[Y] * w1a + tx[X] * w1b + b1c;
        const float g = 0.5f * pre * (1.0f + erff(pre * 0.70710678118654752f));
        hrow[kl] = (_Float16)g;                    // hidT[c][k] (transposed for B frags)
      }
    }
    __syncthreads();

    for (int kb = 0; kb < 288; kb += 32) {
      // A fragments (16x32 f16): two contiguous 8-f16 chunks per lane from global vh
      v16h afrag[2];
#pragma unroll
      for (int mt = 0; mt < 2; ++mt) {
        const _Float16* ap = vh + (row0 + mt * 16) * 576 + kbase + kb + colSel;
        v8h lo = *(const v8h*)ap;
        v8h hi = *(const v8h*)(ap + 16);
        afrag[mt] = __builtin_shufflevector(lo, hi, 0, 1, 2, 3, 4, 5, 6, 7,
                                            8, 9, 10, 11, 12, 13, 14, 15);
      }
      // B fragments (32x16 f16): two 8-K chunks per lane (same K order as A) from LDS hid tile
#pragma unroll
      for (int nt = 0; nt < 4; ++nt) {
        const _Float16* bp = hidT + (nt * 16 + (lane & 15)) * HID_PAD + kb + kgsel;
        v8h lo = *(const v8h*)bp;
        v8h hi = *(const v8h*)(bp + 16);
        v16h bfrag = __builtin_shufflevector(lo, hi, 0, 1, 2, 3, 4, 5, 6, 7,
                                             8, 9, 10, 11, 12, 13, 14, 15);
#pragma unroll
        for (int mt = 0; mt < 2; ++mt) {
          acc[mt][nt] = __builtin_amdgcn_wmma_f32_16x16x32_f16(
              false, afrag[mt], false, bfrag, (short)0, acc[mt][nt], false, false);
        }
      }
    }
  }

  // Spill M accumulators to LDS (C/D layout: VGPR e -> m = e + 8*(lane>=16))
  __syncthreads();
#pragma unroll
  for (int mt = 0; mt < 2; ++mt)
#pragma unroll
    for (int nt = 0; nt < 4; ++nt)
#pragma unroll
      for (int e = 0; e < 8; ++e) {
        const int m = e + ((lane >> 4) << 3);
        Mlds[(wv * 32 + mt * 16 + m) * 64 + nt * 16 + (lane & 15)] = acc[mt][nt][e];
      }
  __syncthreads();

  // Phase 2: thread (b,o); lanes vary o -> M reads broadcast within wave
  const int o = t & 15;
  const int b = t >> 4;
  float dot = 0.f;
  const float4* w2v4 = (const float4*)w2;          // (256,64) row-major
  for (int i = 0; i < 16; ++i) {
    const float4* wrow = w2v4 + (o * 16 + i) * 16;
    const float4* mrow = (const float4*)(Mlds + (b * 16 + i) * 64);
#pragma unroll
    for (int c4 = 0; c4 < 16; ++c4) {
      const float4 a4 = wrow[c4];
      const float4 m4 = mrow[c4];
      dot += m4.x * a4.x + m4.y * a4.y + m4.z * a4.z + m4.w * a4.w;
    }
  }
  float bterm = 0.f;
#pragma unroll
  for (int i = 0; i < 16; ++i) bterm += b2[o * 16 + i] * S[b * 16 + i];

  out[(b * 16 + o) * 576 + hw] = (dot + bterm) * (1.0f / 576.0f) + bias[o];
}

// ---------------------------------------------------------------------------
extern "C" void kernel_launch(void* const* d_in, const int* in_sizes, int n_in,
                              void* d_out, int out_size, void* d_ws, size_t ws_size,
                              hipStream_t stream) {
  (void)in_sizes; (void)n_in; (void)out_size; (void)ws_size;
  const float* v    = (const float*)d_in[0];   // (16,16,24,24)
  const float* w1   = (const float*)d_in[1];   // (64,2)
  const float* b1   = (const float*)d_in[2];   // (64,)
  const float* w2   = (const float*)d_in[3];   // (256,64)
  const float* b2   = (const float*)d_in[4];   // (256,)
  const float* bias = (const float*)d_in[5];   // (16,1,1)
  float* out = (float*)d_out;                  // (16,16,24,24)

  _Float16* vh = (_Float16*)d_ws;                       // 256*576 f16 = 294912 B
  float*    S  = (float*)((char*)d_ws + 294912);        // 256 f32

  prep_kernel<<<1, 256, 0, stream>>>(v, vh, S);
  lin2d_kernel<<<576, 256, 0, stream>>>(vh, S, w1, b1, w2, b2, bias, out);
}
